// LSTMWithLinear_48335561949872
// MI455X (gfx1250) — matrix operand order repaired and checked by reading, not compiled, not run on hardware
//
#include <hip/hip_runtime.h>
#include <hip/hip_bf16.h>

typedef __attribute__((ext_vector_type(16))) __bf16    v16bf;
typedef __attribute__((ext_vector_type(8)))  float     v8f;
typedef __attribute__((ext_vector_type(4)))  unsigned  u32x4;
typedef __attribute__((ext_vector_type(8)))  unsigned  u32x8;

#define USE_TDM 1

#define B_   32
#define T_   512
#define I_   256
#define H1_  512
#define H2_  256
#define F_   8
#define G_   1024
#define TC_  64
#define NCHUNK_ (T_ / TC_)

static __device__ __forceinline__ unsigned short f2bf(float f) {
  union { float f; unsigned u; } v; v.f = f;
  unsigned r = v.u + 0x7FFFu + ((v.u >> 16) & 1u);
  return (unsigned short)(r >> 16);
}
static __device__ __forceinline__ float sigmoidf_(float x) {
  return 1.0f / (1.0f + __expf(-x));
}

union ABfrag { unsigned u[8]; v16bf v; };

// ---------------------------------------------------------------------------
// TDM: issue a 2-D tile load (bf16 elements) into LDS via the Tensor Data
// Mover.  D# group0/group1 per CDNA5 ISA ch.8 (2-D => VADDR2/3 omitted).
//   tile_d0 = tile width in elements (x, contiguous), tile_d1 = rows,
//   stride  = tensor_dim0_stride in elements.
// Wave-level op (EXEC ignored); caller gates to one wave and orders with
// s_wait_tensorcnt + workgroup barrier.
// ---------------------------------------------------------------------------
static __device__ __forceinline__ void tdm_load_2d_bf16(
    unsigned lds_byte_addr, const void* gptr,
    unsigned tile_d0, unsigned tile_d1, unsigned stride_elems) {
  const unsigned long long ga = (unsigned long long)(size_t)gptr;
  const unsigned TD = 1u << 30;  // huge tensor dims: no OOB clipping needed
  u32x4 g0;
  g0[0] = 1u;                                   // count=1, no gather
  g0[1] = lds_byte_addr;                        // lds_addr [63:32]
  g0[2] = (unsigned)ga;                         // global_addr[31:0]
  g0[3] = (unsigned)(ga >> 32) | (2u << 30);    // global_addr[56:32] | type=2
  u32x8 g1;
  g1[0] = 1u << 16;                             // data_size=1 (2 bytes)
  g1[1] = (TD & 0xFFFFu) << 16;                 // tensor_dim0[15:0]
  g1[2] = (TD >> 16) | ((TD & 0xFFFFu) << 16);  // tdim0[31:16] | tdim1[15:0]
  g1[3] = (TD >> 16) | (tile_d0 << 16);         // tdim1[31:16] | tile_dim0
  g1[4] = tile_d1;                              // tile_dim1, tile_dim2=0
  g1[5] = stride_elems;                         // tensor_dim0_stride[31:0]
  g1[6] = 0;                                    // stride[47:32], d1stride lo
  g1[7] = 0;
  asm volatile("tensor_load_to_lds %0, %1" :: "s"(g0), "s"(g1) : "memory");
}

// ---------------------------------------------------------------------------
// f32 -> bf16 conversion (grid-stride)
// ---------------------------------------------------------------------------
__global__ void cvt_bf16_kernel(const float* __restrict__ src,
                                unsigned short* __restrict__ dst, int n) {
  for (int i = blockIdx.x * blockDim.x + threadIdx.x; i < n;
       i += gridDim.x * blockDim.x)
    dst[i] = f2bf(src[i]);
}

__global__ void add_bias_kernel(const float* __restrict__ a,
                                const float* __restrict__ b,
                                float* __restrict__ o, int n) {
  for (int i = blockIdx.x * blockDim.x + threadIdx.x; i < n;
       i += gridDim.x * blockDim.x)
    o[i] = a[i] + b[i];
}

// ---------------------------------------------------------------------------
// Generic bf16 WMMA GEMM:  C[m,n] = act( sum_k A[map(m),k] * W[n,k] + bias[n] )
// Block: 256 threads (8 waves), tile 64(M) x 128(N), k-step 32.
// Wave grid 2x4; each wave computes 2x2 16x16 WMMA tiles.
// A row mapping: elem_off(m) = (m / rpb) * bstride + (m % rpb) * K.
// All launches keep 64-row M-tiles inside one rpb span, so each A tile is a
// uniform-stride 2-D tile -> both A and W tiles staged via TDM DMA.
// Per-head (blockIdx.z): W += f*N*K, bias += f*N, C += f*M*N, A += f*a_fstride.
// ---------------------------------------------------------------------------
template <bool RELU, bool OUTBF16>
__global__ __launch_bounds__(256) void gemm_wmma_kernel(
    const unsigned short* __restrict__ A, const unsigned short* __restrict__ W,
    const float* __restrict__ bias, void* __restrict__ Cv,
    int M, int N, int K, int rpb, long long bstride, long long a_fstride) {
  __shared__ unsigned Atile[64 * 16];    // 64 rows x 32 bf16 (as u32 pairs)
  __shared__ unsigned Wtile[128 * 16];   // 128 rows x 32 bf16

  const int f = blockIdx.z;
  A += (size_t)f * a_fstride;
  W += (size_t)f * (size_t)N * K;
  bias += (size_t)f * N;

  const int t    = threadIdx.x;
  const int wid  = t >> 5, lane = t & 31;
  const int ln   = lane & 15, hi = lane >> 4;
  const int mw   = wid >> 2, nw = wid & 3;

  v8f acc[2][2] = {};

  const int row0 = blockIdx.x * 64;
  const long long aoff0 =
      (long long)(row0 / rpb) * bstride + (long long)(row0 % rpb) * K;
  const unsigned short* abase = A + aoff0;
  const unsigned short* wbase = W + (size_t)(blockIdx.y * 128) * K;

#if !USE_TDM
  const int ar = t >> 2, aq = t & 3;
  const int wr = t >> 1, wq = t & 1;
  const unsigned* arow = (const unsigned*)(abase + (size_t)ar * K);
  const unsigned* wrow = (const unsigned*)(wbase + (size_t)wr * K);
#endif

  for (int k0 = 0; k0 < K; k0 += 32) {
#if USE_TDM
    if (wid == 0) {
      tdm_load_2d_bf16((unsigned)(size_t)&Atile[0], abase + k0, 32, 64, K);
      tdm_load_2d_bf16((unsigned)(size_t)&Wtile[0], wbase + k0, 32, 128, K);
      __builtin_amdgcn_s_wait_tensorcnt(0);
    }
#else
    u32x4 av = *(const u32x4*)(arow + (k0 >> 1) + aq * 4);
    *(u32x4*)&Atile[ar * 16 + aq * 4] = av;
    u32x4 w0 = *(const u32x4*)(wrow + (k0 >> 1) + wq * 8);
    u32x4 w1 = *(const u32x4*)(wrow + (k0 >> 1) + wq * 8 + 4);
    *(u32x4*)&Wtile[wr * 16 + wq * 8]     = w0;
    *(u32x4*)&Wtile[wr * 16 + wq * 8 + 4] = w1;
#endif
    __syncthreads();

    ABfrag a[2], b[2];
#pragma unroll
    for (int mt = 0; mt < 2; ++mt) {  // A 16x32 layout: lane-half holds 16 contiguous K
      const int m = mw * 32 + mt * 16 + ln;
      const int kb2 = hi * 8;
#pragma unroll
      for (int v = 0; v < 4; ++v) a[mt].u[v]     = Atile[m * 16 + kb2 + v];
#pragma unroll
      for (int v = 0; v < 4; ++v) a[mt].u[4 + v] = Atile[m * 16 + kb2 + 4 + v];
    }
#pragma unroll
    for (int nt = 0; nt < 2; ++nt) {  // B 32x16: lane-half holds 16 contiguous K
      const int n = nw * 32 + nt * 16 + ln;
#pragma unroll
      for (int v = 0; v < 8; ++v) b[nt].u[v] = Wtile[n * 16 + hi * 8 + v];
    }
#pragma unroll
    for (int mt = 0; mt < 2; ++mt)
#pragma unroll
      for (int nt = 0; nt < 2; ++nt)
        acc[mt][nt] = __builtin_amdgcn_wmma_f32_16x16x32_bf16(
            false, a[mt].v, false, b[nt].v, (short)0, acc[mt][nt], false, false);
    __syncthreads();
  }

#pragma unroll
  for (int mt = 0; mt < 2; ++mt)
#pragma unroll
    for (int nt = 0; nt < 2; ++nt) {
      const int n_g = blockIdx.y * 128 + nw * 32 + nt * 16 + ln;
      const float bs = bias ? bias[n_g] : 0.0f;
#pragma unroll
      for (int r = 0; r < 8; ++r) {
        const int m_g = row0 + mw * 32 + mt * 16 + r + 8 * hi;
        float val = acc[mt][nt][r] + bs;
        if (RELU) val = fmaxf(val, 0.0f);
        const size_t cidx = ((size_t)f * M + m_g) * N + n_g;
        if (OUTBF16) ((unsigned short*)Cv)[cidx] = f2bf(val);
        else         ((float*)Cv)[cidx] = val;
      }
    }
}

// ---------------------------------------------------------------------------
// Persistent LSTM recurrence for one chunk of TC_ timesteps.
// One workgroup per head (1024 threads = 32 waves). Per step:
//   gates[32x1024] = pre[:,t,:] + h[32x256] @ Whh^T   (bf16 WMMA, f32 acc)
//   elementwise LSTM update, new h written back to LDS as bf16.
// Whh streams from L2 (4 MB total for 8 heads -> L2-resident).
// LDS: 16 KB h (bf16 pairs) + 128 KB gates = 144 KB (<= 320 KB/WGP).
// LAYER==0 writes h chunk (bf16); LAYER==1 fuses the Wa head into y.
// ---------------------------------------------------------------------------
template <int LAYER>
__global__ __launch_bounds__(1024) void lstm_recur_kernel(
    const float* __restrict__ pre,       // [F][B][TC][G] f32
    const unsigned* __restrict__ whh,    // [F][G][H2/2] bf16 pairs
    float* __restrict__ c_state,         // [F][B][H2] f32
    unsigned* __restrict__ h_state,      // [F][B][H2/2] bf16 pairs
    unsigned short* __restrict__ h_out,  // [F][B][TC][H2] bf16 (LAYER==0)
    float* __restrict__ y,               // [B][T][F] f32     (LAYER==1)
    const float* __restrict__ Wa, const float* __restrict__ ba,
    int chunk_idx, int t0) {
  __shared__ unsigned hp[B_ * (H2_ / 2)];  // 16 KB: h as bf16 pairs
  __shared__ float gates[B_ * G_];         // 128 KB

  const int f    = blockIdx.x;
  const int t    = threadIdx.x;
  const int wid  = t >> 5, lane = t & 31;
  const int ln   = lane & 15, hi = lane >> 4;
  const int nb   = wid * 32;               // this wave's 32 gate columns
  const int b    = t >> 5;                 // update-phase batch row

  const unsigned* whhf = whh + (size_t)f * G_ * (H2_ / 2);

  float c8[8], wa8[8];
#pragma unroll
  for (int j = 0; j < 8; ++j) {
    const int u = lane + 32 * j;
    c8[j]  = (chunk_idx == 0) ? 0.0f : c_state[((size_t)f * B_ + b) * H2_ + u];
    wa8[j] = (LAYER == 1) ? Wa[u] : 0.0f;
  }
  for (int i = t; i < B_ * (H2_ / 2); i += 1024)
    hp[i] = (chunk_idx == 0) ? 0u : h_state[(size_t)f * B_ * (H2_ / 2) + i];
  const float ba0 = (LAYER == 1) ? ba[0] : 0.0f;

  for (int tl = 0; tl < TC_; ++tl) {
    __syncthreads();  // h in LDS ready
    // warm next timestep's 128 KB gate-preactivation slab into cache
    if (tl + 1 < TC_)
      __builtin_prefetch(
          &pre[(((size_t)f * B_ + b) * TC_ + (tl + 1)) * G_ + lane * 32], 0, 1);
    // ---- gates = pre + h @ Whh^T ----
    v8f acc[2][2];
#pragma unroll
    for (int mt = 0; mt < 2; ++mt)
#pragma unroll
      for (int nt = 0; nt < 2; ++nt) {
        const int n = nb + nt * 16 + ln;
#pragma unroll
        for (int r = 0; r < 8; ++r) {
          const int m = mt * 16 + r + 8 * hi;
          acc[mt][nt][r] = pre[(((size_t)f * B_ + m) * TC_ + tl) * G_ + n];
        }
      }
    for (int k0 = 0; k0 < H2_; k0 += 32) {
      ABfrag a[2], bf[2];
#pragma unroll
      for (int mt = 0; mt < 2; ++mt) {
        const int m = mt * 16 + ln;
        const int kb2 = (k0 >> 1) + hi * 8;
#pragma unroll
        for (int v = 0; v < 4; ++v) a[mt].u[v]     = hp[m * (H2_ / 2) + kb2 + v];
#pragma unroll
        for (int v = 0; v < 4; ++v) a[mt].u[4 + v] = hp[m * (H2_ / 2) + kb2 + 4 + v];
      }
#pragma unroll
      for (int nt = 0; nt < 2; ++nt) {
        const int n = nb + nt * 16 + ln;
        const unsigned* wr = whhf + (size_t)n * (H2_ / 2) + (k0 >> 1) + hi * 8;
        u32x4 lo = *(const u32x4*)wr;
        u32x4 hi4 = *(const u32x4*)(wr + 4);
#pragma unroll
        for (int v = 0; v < 4; ++v) { bf[nt].u[v] = lo[v]; bf[nt].u[4 + v] = hi4[v]; }
      }
#pragma unroll
      for (int mt = 0; mt < 2; ++mt)
#pragma unroll
        for (int nt = 0; nt < 2; ++nt)
          acc[mt][nt] = __builtin_amdgcn_wmma_f32_16x16x32_bf16(
              false, a[mt].v, false, bf[nt].v, (short)0, acc[mt][nt], false, false);
    }
#pragma unroll
    for (int mt = 0; mt < 2; ++mt)
#pragma unroll
      for (int nt = 0; nt < 2; ++nt) {
        const int n = nb + nt * 16 + ln;
#pragma unroll
        for (int r = 0; r < 8; ++r)
          gates[(mt * 16 + r + 8 * hi) * G_ + n] = acc[mt][nt][r];
      }
    __syncthreads();
    // ---- elementwise LSTM update (gate order i,f,g,o) ----
    float part = 0.0f;
#pragma unroll
    for (int j = 0; j < 8; ++j) {
      const int u = lane + 32 * j;
      const float ig = gates[b * G_ + u];
      const float fg = gates[b * G_ + 256 + u];
      const float gg = gates[b * G_ + 512 + u];
      const float og = gates[b * G_ + 768 + u];
      const float c = sigmoidf_(fg) * c8[j] + sigmoidf_(ig) * tanhf(gg);
      c8[j] = c;
      const float h = sigmoidf_(og) * tanhf(c);
      const unsigned short hb = f2bf(h);
      ((unsigned short*)hp)[b * H2_ + u] = hb;
      if (LAYER == 0)
        h_out[(((size_t)f * B_ + b) * TC_ + tl) * H2_ + u] = hb;
      else
        part += wa8[j] * h;
    }
    if (LAYER == 1) {
#pragma unroll
      for (int s = 16; s > 0; s >>= 1) part += __shfl_down(part, s, 32);
      if (lane == 0) y[((size_t)b * T_ + (t0 + tl)) * F_ + f] = part + ba0;
    }
  }
  __syncthreads();
#pragma unroll
  for (int j = 0; j < 8; ++j)
    c_state[((size_t)f * B_ + b) * H2_ + (lane + 32 * j)] = c8[j];
  for (int i = t; i < B_ * (H2_ / 2); i += 1024)
    h_state[(size_t)f * B_ * (H2_ / 2) + i] = hp[i];
}

// ---------------------------------------------------------------------------
extern "C" void kernel_launch(void* const* d_in, const int* in_sizes, int n_in,
                              void* d_out, int out_size, void* d_ws,
                              size_t ws_size, hipStream_t stream) {
  (void)in_sizes; (void)n_in; (void)out_size; (void)ws_size;
  const float* X    = (const float*)d_in[0];
  const float* W1   = (const float*)d_in[1];
  const float* b1   = (const float*)d_in[2];
  const float* Wih0 = (const float*)d_in[3];
  const float* Whh0 = (const float*)d_in[4];
  const float* bih0 = (const float*)d_in[5];
  const float* bhh0 = (const float*)d_in[6];
  const float* Wih1 = (const float*)d_in[7];
  const float* Whh1 = (const float*)d_in[8];
  const float* bih1 = (const float*)d_in[9];
  const float* bhh1 = (const float*)d_in[10];
  const float* Wa   = (const float*)d_in[11];
  const float* ba   = (const float*)d_in[12];

  char* p = (char*)d_ws;
  auto alloc = [&](size_t bytes) -> void* {
    void* r = (void*)p;
    p += (bytes + 255) & ~(size_t)255;
    return r;
  };
  const int nX    = B_ * T_ * I_;        // 4,194,304
  const int nW1   = H1_ * I_;            // 131,072
  const int nWih0 = F_ * G_ * H1_;       // 4,194,304
  const int nWhh  = F_ * G_ * H2_;       // 2,097,152
  unsigned short* xbf    = (unsigned short*)alloc((size_t)nX * 2);
  unsigned short* w1bf   = (unsigned short*)alloc((size_t)nW1 * 2);
  unsigned short* xhbf   = (unsigned short*)alloc((size_t)B_ * T_ * H1_ * 2);
  unsigned short* wih0bf = (unsigned short*)alloc((size_t)nWih0 * 2);
  unsigned short* whh0bf = (unsigned short*)alloc((size_t)nWhh * 2);
  unsigned short* wih1bf = (unsigned short*)alloc((size_t)nWhh * 2);
  unsigned short* whh1bf = (unsigned short*)alloc((size_t)nWhh * 2);
  float* bsum0 = (float*)alloc((size_t)F_ * G_ * 4);
  float* bsum1 = (float*)alloc((size_t)F_ * G_ * 4);
  float* pre   = (float*)alloc((size_t)F_ * B_ * TC_ * G_ * 4);     // 64 MiB
  unsigned short* h0c = (unsigned short*)alloc((size_t)F_ * B_ * TC_ * H2_ * 2);
  float*    c0s = (float*)alloc((size_t)F_ * B_ * H2_ * 4);
  unsigned* h0s = (unsigned*)alloc((size_t)F_ * B_ * (H2_ / 2) * 4);
  float*    c1s = (float*)alloc((size_t)F_ * B_ * H2_ * 4);
  unsigned* h1s = (unsigned*)alloc((size_t)F_ * B_ * (H2_ / 2) * 4);

  // 1) convert inputs/weights to bf16, fold biases
  cvt_bf16_kernel<<<1024, 256, 0, stream>>>(X, xbf, nX);
  cvt_bf16_kernel<<<512, 256, 0, stream>>>(W1, w1bf, nW1);
  cvt_bf16_kernel<<<1024, 256, 0, stream>>>(Wih0, wih0bf, nWih0);
  cvt_bf16_kernel<<<1024, 256, 0, stream>>>(Whh0, whh0bf, nWhh);
  cvt_bf16_kernel<<<1024, 256, 0, stream>>>(Wih1, wih1bf, nWhh);
  cvt_bf16_kernel<<<1024, 256, 0, stream>>>(Whh1, whh1bf, nWhh);
  add_bias_kernel<<<32, 256, 0, stream>>>(bih0, bhh0, bsum0, F_ * G_);
  add_bias_kernel<<<32, 256, 0, stream>>>(bih1, bhh1, bsum1, F_ * G_);

  // 2) Xh = relu(X @ W1^T + b1), stored bf16  (M=16384, N=512, K=256)
  gemm_wmma_kernel<true, true><<<dim3(256, 4, 1), 256, 0, stream>>>(
      xbf, w1bf, b1, xhbf, B_ * T_, H1_, I_, B_ * T_, 0, 0);

  // 3) chunked 2-layer LSTM over time
  for (int ci = 0; ci < NCHUNK_; ++ci) {
    const int t0 = ci * TC_;
    // layer-0 input GEMM: pre = Xh(chunk) @ Wih0^T + (bih0+bhh0)
    gemm_wmma_kernel<false, false><<<dim3(32, 8, F_), 256, 0, stream>>>(
        xhbf + (size_t)t0 * H1_, wih0bf, bsum0, pre,
        B_ * TC_, G_, H1_, TC_, (long long)T_ * H1_, 0);
    lstm_recur_kernel<0><<<F_, 1024, 0, stream>>>(
        pre, (const unsigned*)whh0bf, c0s, h0s, h0c,
        nullptr, nullptr, nullptr, ci, t0);
    // layer-1 input GEMM: pre = h0(chunk) @ Wih1^T + (bih1+bhh1)
    gemm_wmma_kernel<false, false><<<dim3(32, 8, F_), 256, 0, stream>>>(
        h0c, wih1bf, bsum1, pre,
        B_ * TC_, G_, H2_, B_ * TC_, 0, (long long)B_ * TC_ * H2_);
    lstm_recur_kernel<1><<<F_, 1024, 0, stream>>>(
        pre, (const unsigned*)whh1bf, c1s, h1s, nullptr,
        (float*)d_out, Wa, ba, ci, t0);
  }
}
